// GeometricAttention_70222715290238
// MI455X (gfx1250) — matrix operand order repaired and checked by reading, not compiled, not run on hardware
//
#include <hip/hip_runtime.h>
#include <hip/hip_bf16.h>

typedef __attribute__((ext_vector_type(16))) __bf16 v16bf;
typedef __attribute__((ext_vector_type(8)))  float  v8f;
typedef __attribute__((__vector_size__(16))) int    v4i_t;

#define D_MODEL 1024
#define NUM_HEADS 16
#define HEAD_DIM 64
#define BATCH 2
#define SEQ 2048
#define POS_DIM 64
#define M_ROWS (BATCH * SEQ)   // 4096

// ---- CDNA5 async global->LDS copy (guarded; falls back to 2-phase staging) --
#if defined(__has_builtin)
#if __has_builtin(__builtin_amdgcn_global_load_async_to_lds_b128)
#define ASYNC_LDS 1
#else
#define ASYNC_LDS 0
#endif
#else
#define ASYNC_LDS 0
#endif

#if ASYNC_LDS
#if __has_builtin(__builtin_amdgcn_s_wait_asynccnt)
#define WAIT_ASYNC() __builtin_amdgcn_s_wait_asynccnt(0)
#else
#define WAIT_ASYNC() asm volatile("s_wait_asynccnt 0" ::: "memory")
#endif
__device__ __forceinline__ void async_cp16(const void* g, void* l) {
    __builtin_amdgcn_global_load_async_to_lds_b128(
        (__attribute__((address_space(1))) v4i_t*)
            (__attribute__((address_space(1))) void*)g,
        (__attribute__((address_space(3))) v4i_t*)
            (__attribute__((address_space(3))) void*)l,
        0, 0);
}
#endif

__device__ __forceinline__ unsigned short f32_to_bf16(float f) {
    unsigned u = __float_as_uint(f);
    unsigned r = u + 0x7FFFu + ((u >> 16) & 1u);   // round-to-nearest-even
    return (unsigned short)(r >> 16);
}

union FragB16 {
    uint4 q[2];
    unsigned short u[16];
    v16bf v;
};

// ---------------------------------------------------------------------------
// f32 -> bf16 conversion, 4 elements / thread
// ---------------------------------------------------------------------------
__global__ void geo_cvt_bf16(const float* __restrict__ in,
                             unsigned short* __restrict__ out, int n4) {
    int i = blockIdx.x * 256 + threadIdx.x;
    if (i < n4) {
        float4 f = ((const float4*)in)[i];
        uint2 o;
        o.x = (unsigned)f32_to_bf16(f.x) | ((unsigned)f32_to_bf16(f.y) << 16);
        o.y = (unsigned)f32_to_bf16(f.z) | ((unsigned)f32_to_bf16(f.w) << 16);
        ((uint2*)out)[i] = o;
    }
}

// ---------------------------------------------------------------------------
// f32 [K][N] -> bf16 transposed [N][K] (32x32 LDS tiles, conflict-free)
// ---------------------------------------------------------------------------
__global__ __launch_bounds__(256) void geo_cvt_t_bf16(
        const float* __restrict__ in, unsigned short* __restrict__ out) {
    __shared__ unsigned short tile[32][33];
    const int k0 = blockIdx.y * 32;
    const int n0 = blockIdx.x * 32;
    const int t = threadIdx.x;
#pragma unroll
    for (int i = 0; i < 4; ++i) {
        int e = t + i * 256;
        int r = e >> 5, c = e & 31;
        tile[r][c] = f32_to_bf16(in[(unsigned)(k0 + r) * D_MODEL + n0 + c]);
    }
    __syncthreads();
#pragma unroll
    for (int i = 0; i < 4; ++i) {
        int e = t + i * 256;
        int r = e >> 5, c = e & 31;
        out[(unsigned)(n0 + r) * D_MODEL + k0 + c] = tile[c][r];
    }
}

// ---------------------------------------------------------------------------
// dist[i][j] = ||pos_j - pos_i||
// ---------------------------------------------------------------------------
__global__ __launch_bounds__(256) void geo_dist(const float* __restrict__ pos,
                                                float* __restrict__ dist) {
    __shared__ float pi[POS_DIM];
    const int i = blockIdx.x;
    if (threadIdx.x < POS_DIM) pi[threadIdx.x] = pos[i * POS_DIM + threadIdx.x];
    __syncthreads();
    for (int j = threadIdx.x; j < SEQ; j += 256) {
        const float* pj = pos + (unsigned)j * POS_DIM;
        float a = 0.f;
#pragma unroll
        for (int p = 0; p < POS_DIM; ++p) {
            float d = pj[p] - pi[p];
            a = fmaf(d, d, a);
        }
        dist[(unsigned)i * SEQ + j] = sqrtf(a);
    }
}

// ---------------------------------------------------------------------------
// bf16 GEMM: C[M,N] = scale * (A[M,K] @ BT[N,K]^T + bias[N])
// 128 threads = 4 waves; block tile 64x64; wave tile 16x64; K-step 32.
// ---------------------------------------------------------------------------
#define AST 40   // LDS row stride in ushort (80 B, 16B multiple)

__global__ __launch_bounds__(128) void geo_gemm_bf16(
        const unsigned short* __restrict__ A,
        const unsigned short* __restrict__ BT,   // [N][K] row-major
        const float* __restrict__ bias,
        void* __restrict__ Cout,
        int M, int N, int K, int out_bf16, float scale) {
    __shared__ unsigned short As[64 * AST];
    __shared__ unsigned short Bs[64 * AST];

    const int t     = threadIdx.x;
    const int lane  = t & 31;
    const int wave  = t >> 5;
    const int lhalf = lane >> 4;
    const int lmod  = lane & 15;
    const int m0 = blockIdx.y * 64;
    const int n0 = blockIdx.x * 64;

    // Per-thread staging pointers (2 x 16B chunks per matrix), hoisted.
    const int c0 = t, c1 = t + 128;   // chunk ids; row = c>>2, off = (c&3)*8
    const unsigned short* pa0 = A  + (unsigned)(m0 + (c0 >> 2)) * K + (c0 & 3) * 8;
    const unsigned short* pa1 = A  + (unsigned)(m0 + (c1 >> 2)) * K + (c1 & 3) * 8;
    const unsigned short* pb0 = BT + (unsigned)(n0 + (c0 >> 2)) * K + (c0 & 3) * 8;
    const unsigned short* pb1 = BT + (unsigned)(n0 + (c1 >> 2)) * K + (c1 & 3) * 8;
    unsigned short* la0 = &As[(c0 >> 2) * AST + (c0 & 3) * 8];
    unsigned short* la1 = &As[(c1 >> 2) * AST + (c1 & 3) * 8];
    unsigned short* lb0 = &Bs[(c0 >> 2) * AST + (c0 & 3) * 8];
    unsigned short* lb1 = &Bs[(c1 >> 2) * AST + (c1 & 3) * 8];

    v8f acc[4] = {};
    const int nsteps = K >> 5;
    for (int s = 0; s < nsteps; ++s) {
        __syncthreads();   // previous iteration's LDS reads complete
#if ASYNC_LDS
        async_cp16(pa0, la0);
        async_cp16(pa1, la1);
        async_cp16(pb0, lb0);
        async_cp16(pb1, lb1);
        WAIT_ASYNC();
#else
        uint4 va0 = *(const uint4*)pa0;
        uint4 va1 = *(const uint4*)pa1;
        uint4 vb0 = *(const uint4*)pb0;
        uint4 vb1 = *(const uint4*)pb1;
        *(uint4*)la0 = va0;
        *(uint4*)la1 = va1;
        *(uint4*)lb0 = vb0;
        *(uint4*)lb1 = vb1;
#endif
        if (s + 1 < nsteps) {
            __builtin_prefetch(pa0 + 32, 0, 3);
            __builtin_prefetch(pb0 + 32, 0, 3);
        }
        pa0 += 32; pa1 += 32; pb0 += 32; pb1 += 32;
        __syncthreads();

        // A fragment: m = wave*16 + lmod, k chunks at 8*lhalf, 16+8*lhalf
        FragB16 af;
        {
            const unsigned short* ar = &As[(wave * 16 + lmod) * AST + 8 * lhalf];
            af.q[0] = *(const uint4*)(ar);
            af.q[1] = *(const uint4*)(ar + 16);
        }
#pragma unroll
        for (int tt = 0; tt < 4; ++tt) {
            // B fragment: n-row = tt*16 + lmod, k chunks at 8*lhalf, 16+8*lhalf
            FragB16 bf_;
            const unsigned short* br = &Bs[(tt * 16 + lmod) * AST + 8 * lhalf];
            bf_.q[0] = *(const uint4*)(br);
            bf_.q[1] = *(const uint4*)(br + 16);
            acc[tt] = __builtin_amdgcn_wmma_f32_16x16x32_bf16(
                false, af.v, false, bf_.v, (short)0, acc[tt], false, false);
        }
    }

#pragma unroll
    for (int tt = 0; tt < 4; ++tt) {
        int gn = n0 + tt * 16 + lmod;
        float bv = bias[gn];
#pragma unroll
        for (int r = 0; r < 8; ++r) {
            int gm = m0 + wave * 16 + r + 8 * lhalf;
            float val = (acc[tt][r] + bv) * scale;
            if (out_bf16)
                ((unsigned short*)Cout)[(unsigned)gm * N + gn] = f32_to_bf16(val);
            else
                ((float*)Cout)[(unsigned)gm * N + gn] = val;
        }
    }
}

// ---------------------------------------------------------------------------
// Flash attention with distance bias. One wave per (16-query tile, b, h).
// q already pre-scaled by 1/sqrt(d). scores = q.k - dist; online softmax.
// ---------------------------------------------------------------------------
#define KVST 72   // K/V LDS row stride in ushort (144 B, 16B multiple)
#define PST  40   // P LDS row stride in ushort (80 B)

__global__ __launch_bounds__(32) void geo_attn(
        const unsigned short* __restrict__ Q,
        const unsigned short* __restrict__ Km,
        const unsigned short* __restrict__ Vm,
        const float* __restrict__ dist,
        unsigned short* __restrict__ ctx) {
    __shared__ unsigned short Ks[32 * KVST];
    __shared__ unsigned short Vs[32 * KVST];
    __shared__ unsigned short Ps[16 * PST];

    const int lane  = threadIdx.x;
    const int lhalf = lane >> 4;
    const int lmod  = lane & 15;
    const int q0 = blockIdx.x * 16;
    const int bh = blockIdx.y;
    const int b  = bh >> 4;
    const int h  = bh & 15;

    // q fragments (A-layout)
    FragB16 qf0, qf1;
    {
        const unsigned short* qrow =
            Q + (unsigned)(b * SEQ + q0 + lmod) * D_MODEL + h * HEAD_DIM + 8 * lhalf;
        qf0.q[0] = *(const uint4*)(qrow);
        qf0.q[1] = *(const uint4*)(qrow + 16);
        qf1.q[0] = *(const uint4*)(qrow + 32);
        qf1.q[1] = *(const uint4*)(qrow + 48);
    }

    // Hoisted staging pointers: lane copies K/V row (kb + lane).
    const unsigned short* kp =
        Km + (unsigned)(b * SEQ + lane) * D_MODEL + h * HEAD_DIM;
    const unsigned short* vp =
        Vm + (unsigned)(b * SEQ + lane) * D_MODEL + h * HEAD_DIM;
    unsigned short* kld = &Ks[lane * KVST];
    unsigned short* vld = &Vs[lane * KVST];

    // Distance row pointers (row m = r + 8*lhalf).
    const float* dp[8];
#pragma unroll
    for (int r = 0; r < 8; ++r)
        dp[r] = dist + (unsigned)(q0 + r + 8 * lhalf) * SEQ + lmod;

    float mrun[8], lrun[8];
#pragma unroll
    for (int r = 0; r < 8; ++r) { mrun[r] = -1e30f; lrun[r] = 0.f; }
    v8f acc[4] = {};

    const unsigned short* vsb = &Vs[8 * lhalf * KVST + lmod];

    for (int kb = 0; kb < SEQ; kb += 32) {
        // Stage K/V (32 rows x 64 bf16 each). Single wave: DS pipe in-order.
#if ASYNC_LDS
#pragma unroll
        for (int c = 0; c < 8; ++c) {
            async_cp16(kp + c * 8, kld + c * 8);
            async_cp16(vp + c * 8, vld + c * 8);
        }
        WAIT_ASYNC();
#else
        uint4 rk[8], rv[8];
#pragma unroll
        for (int c = 0; c < 8; ++c) {
            rk[c] = ((const uint4*)kp)[c];
            rv[c] = ((const uint4*)vp)[c];
        }
#pragma unroll
        for (int c = 0; c < 8; ++c) {
            ((uint4*)kld)[c] = rk[c];
            ((uint4*)vld)[c] = rv[c];
        }
#endif
        kp += 32 * D_MODEL;
        vp += 32 * D_MODEL;

        // scores, key subtiles 0 (kb..kb+15) and 1 (kb+16..kb+31)
        v8f c0 = {}, c1 = {};
        FragB16 kf;
        const unsigned short* kr0 = &Ks[lmod * KVST + 8 * lhalf];
        const unsigned short* kr1 = &Ks[(16 + lmod) * KVST + 8 * lhalf];
        kf.q[0] = *(const uint4*)(kr0);
        kf.q[1] = *(const uint4*)(kr0 + 16);
        c0 = __builtin_amdgcn_wmma_f32_16x16x32_bf16(false, qf0.v, false, kf.v,
                                                     (short)0, c0, false, false);
        kf.q[0] = *(const uint4*)(kr0 + 32);
        kf.q[1] = *(const uint4*)(kr0 + 48);
        c0 = __builtin_amdgcn_wmma_f32_16x16x32_bf16(false, qf1.v, false, kf.v,
                                                     (short)0, c0, false, false);
        kf.q[0] = *(const uint4*)(kr1);
        kf.q[1] = *(const uint4*)(kr1 + 16);
        c1 = __builtin_amdgcn_wmma_f32_16x16x32_bf16(false, qf0.v, false, kf.v,
                                                     (short)0, c1, false, false);
        kf.q[0] = *(const uint4*)(kr1 + 32);
        kf.q[1] = *(const uint4*)(kr1 + 48);
        c1 = __builtin_amdgcn_wmma_f32_16x16x32_bf16(false, qf1.v, false, kf.v,
                                                     (short)0, c1, false, false);

        // Online softmax (row m = r + 8*lhalf spans 16 lanes).
        float corr[8], p0[8], p1[8];
#pragma unroll
        for (int r = 0; r < 8; ++r) {
            float s0 = c0[r] - dp[r][kb];
            float s1 = c1[r] - dp[r][kb + 16];
            float tm = fmaxf(s0, s1);
#pragma unroll
            for (int msk = 1; msk < 16; msk <<= 1)
                tm = fmaxf(tm, __shfl_xor(tm, msk, 32));
            float nm = fmaxf(mrun[r], tm);
            float cr = __expf(mrun[r] - nm);
            mrun[r] = nm;
            float e0 = __expf(s0 - nm);
            float e1 = __expf(s1 - nm);
            float sm = e0 + e1;
#pragma unroll
            for (int msk = 1; msk < 16; msk <<= 1)
                sm += __shfl_xor(sm, msk, 32);
            lrun[r] = lrun[r] * cr + sm;
            corr[r] = cr;
            p0[r] = e0;
            p1[r] = e1;
        }
#pragma unroll
        for (int tt = 0; tt < 4; ++tt)
#pragma unroll
            for (int r = 0; r < 8; ++r) acc[tt][r] *= corr[r];

        // P (C-layout) -> LDS [m][k] -> reload in A-layout.
#pragma unroll
        for (int r = 0; r < 8; ++r) {
            int m = r + 8 * lhalf;
            Ps[m * PST + lmod]      = f32_to_bf16(p0[r]);
            Ps[m * PST + 16 + lmod] = f32_to_bf16(p1[r]);
        }
        FragB16 pf;
        pf.q[0] = *(const uint4*)(&Ps[lmod * PST + 8 * lhalf]);
        pf.q[1] = *(const uint4*)(&Ps[lmod * PST + 16 + 8 * lhalf]);

        // ctx += P @ V ; V fragment (B-layout): k = 8*lhalf + e + 8*(e>>3), n = tt*16+lmod
#pragma unroll
        for (int tt = 0; tt < 4; ++tt) {
            FragB16 vf;
#pragma unroll
            for (int e = 0; e < 16; ++e)
                vf.u[e] = vsb[(e + (e >> 3) * 8) * KVST + tt * 16];
            acc[tt] = __builtin_amdgcn_wmma_f32_16x16x32_bf16(
                false, pf.v, false, vf.v, (short)0, acc[tt], false, false);
        }
    }

    // Normalize and store context (bf16).
#pragma unroll
    for (int tt = 0; tt < 4; ++tt) {
#pragma unroll
        for (int r = 0; r < 8; ++r) {
            int gm = q0 + r + 8 * lhalf;
            float val = acc[tt][r] / lrun[r];
            ctx[(unsigned)(b * SEQ + gm) * D_MODEL + h * HEAD_DIM + tt * 16 + lmod] =
                f32_to_bf16(val);
        }
    }
}

// ---------------------------------------------------------------------------
// Host-side launcher
// ---------------------------------------------------------------------------
extern "C" void kernel_launch(void* const* d_in, const int* in_sizes, int n_in,
                              void* d_out, int out_size, void* d_ws, size_t ws_size,
                              hipStream_t stream) {
    const float* x   = (const float*)d_in[0];
    const float* pos = (const float*)d_in[1];
    const float* Wq  = (const float*)d_in[2];
    const float* bq  = (const float*)d_in[3];
    const float* Wk  = (const float*)d_in[4];
    const float* bk  = (const float*)d_in[5];
    const float* Wv  = (const float*)d_in[6];
    const float* bv  = (const float*)d_in[7];
    const float* Wo  = (const float*)d_in[8];
    const float* bo  = (const float*)d_in[9];
    float* out = (float*)d_out;

    char* ws = (char*)d_ws;
    const size_t XN = (size_t)M_ROWS * D_MODEL;
    unsigned short* xb   = (unsigned short*)(ws + 0);            // 8 MiB
    unsigned short* wqb  = (unsigned short*)(ws + (8u << 20));   // 2 MiB each (transposed)
    unsigned short* wkb  = (unsigned short*)(ws + (10u << 20));
    unsigned short* wvb  = (unsigned short*)(ws + (12u << 20));
    unsigned short* wob  = (unsigned short*)(ws + (14u << 20));
    unsigned short* qb   = (unsigned short*)(ws + (16u << 20));  // 8 MiB each
    unsigned short* kb   = (unsigned short*)(ws + (24u << 20));
    unsigned short* vb   = (unsigned short*)(ws + (32u << 20));
    unsigned short* ctxb = (unsigned short*)(ws + (40u << 20));
    float*          dst  = (float*)(ws + (48u << 20));           // 16 MiB

    // 1) Convert x (straight) and weights (convert + transpose to [N][K]).
    {
        int n4 = (int)(XN / 4);
        geo_cvt_bf16<<<(n4 + 255) / 256, 256, 0, stream>>>(x, xb, n4);
        dim3 tg(D_MODEL / 32, D_MODEL / 32);
        geo_cvt_t_bf16<<<tg, 256, 0, stream>>>(Wq, wqb);
        geo_cvt_t_bf16<<<tg, 256, 0, stream>>>(Wk, wkb);
        geo_cvt_t_bf16<<<tg, 256, 0, stream>>>(Wv, wvb);
        geo_cvt_t_bf16<<<tg, 256, 0, stream>>>(Wo, wob);
    }

    // 2) Distance table.
    geo_dist<<<SEQ, 256, 0, stream>>>(pos, dst);

    // 3) QKV projections (bf16 out; q pre-scaled by 1/sqrt(HEAD_DIM) = 0.125).
    dim3 ggrid(D_MODEL / 64, M_ROWS / 64);
    geo_gemm_bf16<<<ggrid, 128, 0, stream>>>(xb, wqb, bq, qb,
                                             M_ROWS, D_MODEL, D_MODEL, 1, 0.125f);
    geo_gemm_bf16<<<ggrid, 128, 0, stream>>>(xb, wkb, bk, kb,
                                             M_ROWS, D_MODEL, D_MODEL, 1, 1.0f);
    geo_gemm_bf16<<<ggrid, 128, 0, stream>>>(xb, wvb, bv, vb,
                                             M_ROWS, D_MODEL, D_MODEL, 1, 1.0f);

    // 4) Distance-biased flash attention.
    geo_attn<<<dim3(SEQ / 16, BATCH * NUM_HEADS), 32, 0, stream>>>(
        qb, kb, vb, dst, ctxb);

    // 5) Output projection (f32 out).
    geo_gemm_bf16<<<ggrid, 128, 0, stream>>>(ctxb, wob, bo, out,
                                             M_ROWS, D_MODEL, D_MODEL, 0, 1.0f);
}
